// align_net_13838384627984
// MI455X (gfx1250) — matrix operand order repaired and checked by reading, not compiled, not run on hardware
//
#include <hip/hip_runtime.h>
#include <hip/hip_bf16.h>

typedef __bf16 bf16;
typedef __attribute__((ext_vector_type(16))) __bf16 v16bf;
typedef __attribute__((ext_vector_type(8)))  float  v8f;

#define HW  256
#define PW  258          // padded width/height (1px halo)
#define PP  (PW * PW)    // padded pixels
#define NFC 64
#define OMC 256          // 216 channels padded to 16 co-blocks (group of 4)

#define EP_NONE 0
#define EP_RELU 1
#define EP_ADD  2

__device__ __forceinline__ float bf2f(unsigned short u) {
    union { unsigned int i; float f; } c; c.i = ((unsigned int)u) << 16; return c.f;
}
__device__ __forceinline__ unsigned short f2bf(float f) {
    union { float f; unsigned int i; } c; c.f = f;
    unsigned int u = c.i;
    return (unsigned short)((u + 0x7fffu + ((u >> 16) & 1u)) >> 16);
}

// CDNA5 async global->LDS copy (ASYNCcnt-tracked).  lds_off is the byte
// offset within the workgroup LDS allocation (dynamic LDS starts at 0).
__device__ __forceinline__ void async_copy_b128(unsigned int lds_off, const void* gptr) {
    asm volatile("global_load_async_to_lds_b128 %0, %1, off"
                 :: "v"(lds_off), "v"(gptr) : "memory");
}
__device__ __forceinline__ void wait_async0() {
    asm volatile("s_wait_asynccnt 0x0" ::: "memory");
}

// ---------------------------------------------------------------------------
// Zero an arena (16B granules), grid-stride.
// ---------------------------------------------------------------------------
__global__ void zero_buf(uint4* __restrict__ p, long n16)
{
    long i = (long)blockIdx.x * blockDim.x + threadIdx.x;
    long stride = (long)gridDim.x * blockDim.x;
    uint4 z; z.x = 0; z.y = 0; z.z = 0; z.w = 0;
    for (; i < n16; i += stride) p[i] = z;
}

// ---------------------------------------------------------------------------
// Repack conv weights (fp32 OIHW) into WMMA A-fragment order, bf16.
// Layout: [cb][kc][lane][e]  (16 bf16 = 32B contiguous per lane).
// mode 0 (plain conv): k = tap*Ci + ci
// mode 1 (dcn):        k = (g*9 + tap)*8 + c, ci = g*8 + c
// ---------------------------------------------------------------------------
__global__ void pack_weights(const float* __restrict__ w, unsigned short* __restrict__ out,
                             int Co, int Ci, int cbCount, int kcCount, int mode)
{
    int idx = blockIdx.x * 256 + threadIdx.x;
    int total = cbCount * kcCount * 32 * 16;
    if (idx >= total) return;
    int e    = idx & 15;
    int lane = (idx >> 4) & 31;
    int rest = idx >> 9;
    int kc   = rest % kcCount;
    int cb   = rest / kcCount;
    int m  = lane & 15;
    int hi = lane >> 4;
    int kloc = hi * 16 + e;
    int k  = kc * 32 + kloc;
    int co = cb * 16 + m;
    float v = 0.f;
    int Kt = kcCount * 32;
    if (co < Co && k < Kt) {
        int ci, tap;
        if (mode == 0) { tap = k / Ci; ci = k % Ci; }
        else { int c = k & 7; int t2 = k >> 3; tap = t2 % 9; int g = t2 / 9; ci = g * 8 + c; }
        int ky = tap / 3, kx = tap % 3;
        v = w[((co * Ci + ci) * 3 + ky) * 3 + kx];
    }
    out[idx] = f2bf(v);
}

// ---------------------------------------------------------------------------
// conv_first: 3 -> 64 channels, fused ReLU, NCHW fp32 in -> padded CL bf16
// ---------------------------------------------------------------------------
__global__ void conv_first_kernel(const float* __restrict__ x,
                                  const float* __restrict__ w,
                                  const float* __restrict__ b,
                                  unsigned short* __restrict__ feat)
{
    int p = blockIdx.x * blockDim.x + threadIdx.x;
    int total = 3 * HW * HW;
    if (p >= total) return;
    int f   = p / (HW * HW);
    int rem = p % (HW * HW);
    int y = rem / HW, xo = rem % HW;
    const float* xf = x + (size_t)f * 3 * HW * HW;
    float in[27];
    #pragma unroll
    for (int ci = 0; ci < 3; ci++)
        #pragma unroll
        for (int t = 0; t < 9; t++) {
            int yy = y + t / 3 - 1, xx = xo + t % 3 - 1;
            in[ci * 9 + t] = (yy >= 0 && yy < HW && xx >= 0 && xx < HW)
                           ? xf[(ci * HW + yy) * HW + xx] : 0.f;
        }
    unsigned short* o = feat + ((size_t)f * PP + (size_t)(y + 1) * PW + (xo + 1)) * NFC;
    for (int co = 0; co < NFC; co++) {
        float s = b[co];
        #pragma unroll
        for (int j = 0; j < 27; j++) s += w[co * 27 + j] * in[j];
        o[co] = f2bf(fmaxf(s, 0.f));
    }
}

// ---------------------------------------------------------------------------
// 3x3 conv, Cin=64, implicit GEMM on WMMA bf16 (K=576, 18 chunks of 32).
// act: padded (PW,PW,64) bf16.  out: padded (PW,PW,Cpad) bf16, interior only.
// Weights async-staged to LDS; co-blocks in compile-time groups of 4 so each
// activation B-fragment is loaded once per 4 unconditional WMMAs.
// cbCount must be a multiple of 4.
// ---------------------------------------------------------------------------
__global__ void __launch_bounds__(256)
conv3x3_wmma(const unsigned short* __restrict__ act,
             const unsigned short* __restrict__ wpk,
             const float* __restrict__ bias,
             unsigned short* __restrict__ out,
             const unsigned short* __restrict__ res,
             int Co, int Cpad, int cbCount, int mode)
{
    extern __shared__ unsigned short lw[];   // cbCount*18*512 bf16
    const int y    = blockIdx.x;
    const int lane = threadIdx.x & 31;
    const int wv   = threadIdx.x >> 5;
    const int m15  = lane & 15;
    const int hi   = (lane >> 4) & 1;

    // stage packed weights into LDS via async copies
    const int totalB128 = cbCount * 1152;    // cbCount*18*512*2 / 16
    for (int i = threadIdx.x; i < totalB128; i += 256)
        async_copy_b128((unsigned int)(i * 16), (const char*)wpk + (size_t)i * 16);
    wait_async0();
    __syncthreads();

    for (int s = 0; s < 2; ++s) {
        int px0 = (wv * 2 + s) * 16;
        for (int cb0 = 0; cb0 < cbCount; cb0 += 4) {
            v8f acc[4];
            #pragma unroll
            for (int j = 0; j < 4; j++)
                #pragma unroll
                for (int i = 0; i < 8; i++) acc[j][i] = 0.f;
            #pragma unroll 3
            for (int kc = 0; kc < 18; ++kc) {
                int tap = kc >> 1;
                int ky = tap / 3, kx = tap % 3;        // 0..2, halo handles -1
                int ci0 = (kc & 1) * 32 + hi * 16;
                int pix = (y + ky) * PW + (px0 + m15 + kx);
                v16bf bfr = *(const v16bf*)(act + ((size_t)pix * NFC + ci0));
                #pragma unroll
                for (int j = 0; j < 4; ++j) {
                    v16bf a = *(const v16bf*)(lw + ((size_t)((cb0 + j) * 18 + kc) * 32 + lane) * 16);
                    acc[j] = __builtin_amdgcn_wmma_f32_16x16x32_bf16(
                                 false, a, false, bfr, (short)0, acc[j], false, false);
                }
            }
            int xo   = px0 + m15;
            int pixo = (y + 1) * PW + (xo + 1);
            #pragma unroll
            for (int j = 0; j < 4; ++j) {
                int co0 = (cb0 + j) * 16 + hi * 8;
                float r8[8];
                #pragma unroll
                for (int i = 0; i < 8; i++)
                    r8[i] = acc[j][i] + ((co0 + i < Co) ? bias[co0 + i] : 0.f);
                if (mode == EP_ADD) {
                    const unsigned short* rp = res + ((size_t)pixo * Cpad + co0);
                    #pragma unroll
                    for (int i = 0; i < 8; i++) r8[i] += bf2f(rp[i]);
                } else if (mode == EP_RELU) {
                    #pragma unroll
                    for (int i = 0; i < 8; i++) r8[i] = fmaxf(r8[i], 0.f);
                }
                unsigned short t[8];
                #pragma unroll
                for (int i = 0; i < 8; i++) t[i] = f2bf(r8[i]);
                uint4 pk;
                pk.x = t[0] | (t[1] << 16); pk.y = t[2] | (t[3] << 16);
                pk.z = t[4] | (t[5] << 16); pk.w = t[6] | (t[7] << 16);
                *(uint4*)(out + ((size_t)pixo * Cpad + co0)) = pk;
            }
        }
    }
}

// ---------------------------------------------------------------------------
// Modulated deformable conv: bilinear gather -> LDS (bf16 [pix][k=576]) ->
// WMMA GEMM K=576.  Weights async-staged to LDS, overlapped with sampling.
// LDS: [0,147456) = 8 waves * 16px * 576 bf16 vals; [147456,221184) weights.
// ---------------------------------------------------------------------------
#define MD_WOFF 147456
__global__ void __launch_bounds__(256)
mdconv_wmma(const unsigned short* __restrict__ xact,   // padded (PW,PW,64)
            const unsigned short* __restrict__ om,     // padded (PW,PW,256)
            const unsigned short* __restrict__ wpk,
            const float* __restrict__ bias,
            unsigned short* __restrict__ outBf,        // padded (PW,PW,64)
            float* __restrict__ outF,                  // NCHW fp32 (final)
            int finalOut)
{
    extern __shared__ unsigned short lds[];
    const int y    = blockIdx.x;
    const int lane = threadIdx.x & 31;
    const int wv   = threadIdx.x >> 5;
    unsigned short* lv = lds + (size_t)wv * 16 * 576;
    unsigned short* lw = lds + MD_WOFF / 2;
    const int m15 = lane & 15;
    const int hi  = (lane >> 4) & 1;

    // async-stage packed deform weights (73728 B) while we sample
    for (int i = threadIdx.x; i < 4608; i += 256)
        async_copy_b128((unsigned int)(MD_WOFF + i * 16), (const char*)wpk + (size_t)i * 16);

    for (int s = 0; s < 2; ++s) {
        int px0 = (wv * 2 + s) * 16;
        int xp  = px0 + m15;
        const unsigned short* omp = om + ((size_t)(y + 1) * PW + (xp + 1)) * OMC;

        // ---- sampling: 72 taps, 2 taps in flight across the wave ----
        for (int t = hi; t < 72; t += 2) {
            int g = t / 9, k9 = t % 9;
            int ky = k9 / 3 - 1, kx = k9 % 3 - 1;
            float offy = bf2f(omp[t]);
            float offx = bf2f(omp[72 + t]);
            float ml   = bf2f(omp[144 + t]);
            float msk  = 1.0f / (1.0f + __expf(-ml));
            float py  = (float)(y + ky) + offy;
            float pxx = (float)(xp + kx) + offx;
            float y0 = floorf(py), x0 = floorf(pxx);
            float ly = py - y0,  lx = pxx - x0;
            float hy = 1.f - ly, hx = 1.f - lx;
            int yi = (int)y0, xi = (int)x0;
            float v[8];
            #pragma unroll
            for (int i = 0; i < 8; i++) v[i] = 0.f;
            #pragma unroll
            for (int n = 0; n < 4; ++n) {
                int dy = n >> 1, dx = n & 1;
                int yn = yi + dy, xn = xi + dx;
                float wgt = (dy ? ly : hy) * (dx ? lx : hx);
                if (yn >= 0 && yn < HW && xn >= 0 && xn < HW) {
                    uint4 q = *(const uint4*)(xact + (((size_t)(yn + 1) * PW + (xn + 1)) * NFC + g * 8));
                    const unsigned short* qs = (const unsigned short*)&q;
                    #pragma unroll
                    for (int c = 0; c < 8; c++) v[c] += wgt * bf2f(qs[c]);
                }
            }
            unsigned short tb[8];
            #pragma unroll
            for (int c = 0; c < 8; c++) tb[c] = f2bf(v[c] * msk);
            uint4 pk;
            pk.x = tb[0] | (tb[1] << 16); pk.y = tb[2] | (tb[3] << 16);
            pk.z = tb[4] | (tb[5] << 16); pk.w = tb[6] | (tb[7] << 16);
            *(uint4*)(lv + ((size_t)m15 * 576 + t * 8)) = pk;
        }
        wait_async0();
        __syncthreads();

        // ---- GEMM: out[64 x 16px] = W[64 x 576] * val[576 x 16px] ----
        {
            v8f acc[4];
            #pragma unroll
            for (int j = 0; j < 4; j++)
                #pragma unroll
                for (int i = 0; i < 8; i++) acc[j][i] = 0.f;
            #pragma unroll
            for (int kc = 0; kc < 18; ++kc) {
                int kbase = kc * 32 + hi * 16;
                v16bf bfr = *(const v16bf*)(lv + ((size_t)m15 * 576 + kbase));
                #pragma unroll
                for (int j = 0; j < 4; ++j) {
                    v16bf a = *(const v16bf*)(lw + ((size_t)(j * 18 + kc) * 32 + lane) * 16);
                    acc[j] = __builtin_amdgcn_wmma_f32_16x16x32_bf16(
                                 false, a, false, bfr, (short)0, acc[j], false, false);
                }
            }
            int xo = px0 + m15;
            #pragma unroll
            for (int j = 0; j < 4; ++j) {
                int co0 = j * 16 + hi * 8;
                if (finalOut) {
                    #pragma unroll
                    for (int i = 0; i < 8; i++) {
                        float r = fmaxf(acc[j][i] + bias[co0 + i], 0.f);
                        outF[((size_t)(co0 + i) * HW + y) * HW + xo] = r;
                    }
                } else {
                    unsigned short t[8];
                    #pragma unroll
                    for (int i = 0; i < 8; i++) t[i] = f2bf(acc[j][i] + bias[co0 + i]);
                    uint4 pk;
                    pk.x = t[0] | (t[1] << 16); pk.y = t[2] | (t[3] << 16);
                    pk.z = t[4] | (t[5] << 16); pk.w = t[6] | (t[7] << 16);
                    *(uint4*)(outBf + (((size_t)(y + 1) * PW + (xo + 1)) * NFC + co0)) = pk;
                }
            }
        }
        __syncthreads();
    }
}

// ---------------------------------------------------------------------------
// Orchestration
// ---------------------------------------------------------------------------
extern "C" void kernel_launch(void* const* d_in, const int* in_sizes, int n_in,
                              void* d_out, int out_size, void* d_ws, size_t ws_size,
                              hipStream_t stream)
{
    (void)in_sizes; (void)n_in; (void)out_size; (void)ws_size;
    const float* x    = (const float*)d_in[0];
    const float* cf_w = (const float*)d_in[1];
    const float* cf_b = (const float*)d_in[2];

    // allow big dynamic LDS (WGP has 320 KB)
    (void)hipFuncSetAttribute((const void*)conv3x3_wmma,
                              hipFuncAttributeMaxDynamicSharedMemorySize, 16 * 18 * 512 * 2);
    (void)hipFuncSetAttribute((const void*)mdconv_wmma,
                              hipFuncAttributeMaxDynamicSharedMemorySize, MD_WOFF + 73728);

    char* ws = (char*)d_ws;
    size_t off = 0;
    auto carve = [&](size_t bytes) -> char* {
        char* p = ws + off; off += (bytes + 255) & ~(size_t)255; return p;
    };
    const size_t actB = (size_t)PP * NFC * 2;          // padded bf16 feature map
    unsigned short* FEAT = (unsigned short*)carve(3 * actB);
    unsigned short* U0 = (unsigned short*)carve(actB);
    unsigned short* U1 = (unsigned short*)carve(actB);
    unsigned short* U2 = (unsigned short*)carve(actB);
    unsigned short* U3 = (unsigned short*)carve(actB);
    unsigned short* OM = (unsigned short*)carve((size_t)PP * OMC * 2);
    size_t zeroBytes = off;                            // zero whole activation arena

    const size_t pk64  = (size_t)4  * 18 * 32 * 16;
    const size_t pk256 = (size_t)16 * 18 * 32 * 16;
    unsigned short* PK_RES[10];
    unsigned short* PK_OFF[4];
    unsigned short* PK_OM[8];
    unsigned short* PK_DW[8];
    for (int i = 0; i < 10; i++) PK_RES[i] = (unsigned short*)carve(pk64 * 2);
    for (int i = 0; i < 4;  i++) PK_OFF[i] = (unsigned short*)carve(pk64 * 2);
    for (int i = 0; i < 8;  i++) PK_OM[i]  = (unsigned short*)carve(pk256 * 2);
    for (int i = 0; i < 8;  i++) PK_DW[i]  = (unsigned short*)carve(pk64 * 2);

    // zero activation arena (halos must be 0; interiors are overwritten)
    zero_buf<<<2048, 256, 0, stream>>>((uint4*)ws, (long)(zeroBytes / 16));

    auto pack = [&](const float* w, unsigned short* dst, int Co, int cbC, int mode) {
        int total = cbC * 18 * 32 * 16;
        pack_weights<<<(total + 255) / 256, 256, 0, stream>>>(w, dst, Co, NFC, cbC, 18, mode);
    };
    for (int rb = 0; rb < 5; rb++) {
        pack((const float*)d_in[3 + rb * 4 + 0], PK_RES[rb * 2 + 0], 64, 4, 0);
        pack((const float*)d_in[3 + rb * 4 + 2], PK_RES[rb * 2 + 1], 64, 4, 0);
    }
    for (int j = 0; j < 8; j++) {
        pack((const float*)d_in[23 + j * 4 + 0], PK_OM[j], 216, 16, 0);
        pack((const float*)d_in[23 + j * 4 + 2], PK_DW[j], 64, 4, 1);
    }
    for (int j = 0; j < 4; j++)
        pack((const float*)d_in[55 + j * 2 + 0], PK_OFF[j], 64, 4, 0);

    auto conv = [&](const unsigned short* act, const unsigned short* pk, const float* bias,
                    unsigned short* out, const unsigned short* res,
                    int Co, int Cpad, int cbC, int mode) {
        size_t sh = (size_t)cbC * 18 * 512 * 2;
        conv3x3_wmma<<<HW, 256, sh, stream>>>(act, pk, bias, out, res, Co, Cpad, cbC, mode);
    };
    auto mdconv = [&](const unsigned short* xa, const unsigned short* omb,
                      const unsigned short* pk, const float* bias,
                      unsigned short* oBf, float* oF, int fin) {
        mdconv_wmma<<<HW, 256, MD_WOFF + 73728, stream>>>(xa, omb, pk, bias, oBf, oF, fin);
    };

    conv_first_kernel<<<(3 * HW * HW + 255) / 256, 256, 0, stream>>>(x, cf_w, cf_b, FEAT);

    for (int f = 0; f < 3; f++) {
        unsigned short* F = FEAT + (size_t)f * PP * NFC;
        for (int rb = 0; rb < 5; rb++) {
            const float* b1 = (const float*)d_in[3 + rb * 4 + 1];
            const float* b2 = (const float*)d_in[3 + rb * 4 + 3];
            conv(F,  PK_RES[rb * 2 + 0], b1, U0, nullptr, 64, 64, 4, EP_RELU);
            conv(U0, PK_RES[rb * 2 + 1], b2, F,  F,       64, 64, 4, EP_ADD);
        }
    }

    unsigned short* REF = FEAT + (size_t)1 * PP * NFC;
    for (int i = 0; i < 3; i++) {
        unsigned short* S = FEAT + (size_t)i * PP * NFC;
        float* outp = (float*)d_out + (size_t)i * NFC * HW * HW;
        const float* omB[8]; const float* dwB[8];
        for (int j = 0; j < 8; j++) {
            omB[j] = (const float*)d_in[23 + j * 4 + 1];
            dwB[j] = (const float*)d_in[23 + j * 4 + 3];
        }
        conv(REF, PK_OM[0], omB[0], OM, nullptr, 216, OMC, 16, EP_NONE);
        mdconv(S, OM, PK_DW[0], dwB[0], U0, nullptr, 0);
        conv(U0, PK_OFF[0], (const float*)d_in[56], U1, nullptr, 64, 64, 4, EP_RELU);
        conv(U1, PK_OM[1], omB[1], OM, nullptr, 216, OMC, 16, EP_NONE);
        mdconv(S, OM, PK_DW[1], dwB[1], U2, nullptr, 0);
        conv(REF, PK_OM[2], omB[2], OM, nullptr, 216, OMC, 16, EP_NONE);
        mdconv(U2, OM, PK_DW[2], dwB[2], U0, nullptr, 0);
        conv(U0, PK_OFF[1], (const float*)d_in[58], U1, nullptr, 64, 64, 4, EP_RELU);
        conv(U1, PK_OM[3], omB[3], OM, nullptr, 216, OMC, 16, EP_NONE);
        mdconv(U2, OM, PK_DW[3], dwB[3], U3, nullptr, 0);
        conv(REF, PK_OM[4], omB[4], OM, nullptr, 216, OMC, 16, EP_NONE);
        mdconv(U3, OM, PK_DW[4], dwB[4], U0, nullptr, 0);
        conv(U0, PK_OFF[2], (const float*)d_in[60], U1, nullptr, 64, 64, 4, EP_RELU);
        conv(U1, PK_OM[5], omB[5], OM, nullptr, 216, OMC, 16, EP_NONE);
        mdconv(U3, OM, PK_DW[5], dwB[5], U2, nullptr, 0);
        conv(REF, PK_OM[6], omB[6], OM, nullptr, 216, OMC, 16, EP_NONE);
        mdconv(U2, OM, PK_DW[6], dwB[6], U0, nullptr, 0);
        conv(U0, PK_OFF[3], (const float*)d_in[62], U1, nullptr, 64, 64, 4, EP_RELU);
        conv(U1, PK_OM[7], omB[7], OM, nullptr, 216, OMC, 16, EP_NONE);
        mdconv(U2, OM, PK_DW[7], dwB[7], nullptr, outp, 1);
    }
}
